// Model_30812095381599
// MI455X (gfx1250) — compile-verified, matrix operand order checked by reading, not run on hardware
//
#include <hip/hip_runtime.h>
#include <hip/hip_bf16.h>
#include <stdint.h>

// Problem dims
#define BB    128
#define SEQ   720
#define PRED  336
#define ENCD  321
#define DD    512
#define GG    2048          // 4*D gate columns
#define NWG   32            // persistent workgroups; each owns 16 h-columns
#define KTH   16            // 512/32 k-tiles for h
#define KTX   11            // 352/32 k-tiles for x (321 padded to 352)
#define NTY   21            // 336/16 n-tiles for yy (321 padded to 336)
#define H1    (8*KTH*512)   // bf16 elems per h A-frag buffer (one parity) = 65536
#define X1    (8*KTX*512)   // bf16 elems per x A-frag buffer (one parity) = 45056

typedef __attribute__((ext_vector_type(16))) __bf16 v16bf;
typedef __attribute__((ext_vector_type(8)))  float  v8f;

// Workspace layout (bytes, 256-aligned)
#define O_CTRL   0
#define O_HBUF   256
#define O_XBUF   (O_HBUF + 2*H1*2)            // 262400
#define O_WENCH  (O_XBUF + 2*X1*2)            // 442624
#define O_WENCX  (O_WENCH + 128*KTH*512*2)    // 2539776
#define O_WDECH  (O_WENCX + 128*KTX*512*2)    // 3981568
#define O_WP     (O_WDECH + 128*KTH*512*2)    // 6078720
#define O_BENC   (O_WP + NTY*KTH*512*2)       // 6422784
#define O_BDEC   (O_BENC + GG*4)              // 6430976
#define O_BPRD   (O_BDEC + GG*4)              // 6439168

#define SMEM_BYTES ((4*KTH*512 + 4*KTX*512 + 4*KTH*512 + KTH*512)*2)  // 192512

// Fragment index helper (ISA 7.12.2 16-bit A/B layout):
// lane L holds row/col (L&15); half-index p maps to k-offset:
__device__ __forceinline__ int koff_of(int lane, int p) {
  return (p & 7) + ((p >> 3) << 4) + ((lane >> 4) << 3);
}
__device__ __forceinline__ float sigf(float x) { return 1.0f / (1.0f + __expf(-x)); }

__device__ __forceinline__ v8f wmma_bf16(v16bf a, v16bf b, v8f c) {
  return __builtin_amdgcn_wmma_f32_16x16x32_bf16(false, a, false, b, (short)0, c, false, false);
}

// -------- async global -> LDS copy (Tensor/async path, ASYNCcnt) --------
__device__ __forceinline__ void async_g2l(void* ldst, const void* gsrc, int bytes) {
  unsigned ldsbase = (unsigned)(uintptr_t)ldst;              // low 32 bits = LDS offset
  unsigned long long gbase = (unsigned long long)(uintptr_t)gsrc;
  for (int off = (int)threadIdx.x * 16; off < bytes; off += 256 * 16) {
    unsigned l = ldsbase + (unsigned)off;
    asm volatile("global_load_async_to_lds_b128 %0, %1, %2"
                 :: "v"(l), "v"(off), "s"(gbase) : "memory");
  }
}
__device__ __forceinline__ void wait_async_all() {
  asm volatile("s_wait_asynccnt 0" ::: "memory");
}

// -------- device-wide sense barrier --------
__device__ __forceinline__ void grid_sync(unsigned* cnt, unsigned* gen, unsigned& mygen) {
  __builtin_amdgcn_fence(__ATOMIC_RELEASE, "agent");   // make h/x stores visible
  __syncthreads();
  if (threadIdx.x == 0) {
    unsigned prev = __hip_atomic_fetch_add(cnt, 1u, __ATOMIC_RELAXED, __HIP_MEMORY_SCOPE_AGENT);
    if (prev == mygen * NWG + (NWG - 1)) {
      __hip_atomic_fetch_add(gen, 1u, __ATOMIC_RELEASE, __HIP_MEMORY_SCOPE_AGENT);
    } else {
      while (__hip_atomic_load(gen, __ATOMIC_RELAXED, __HIP_MEMORY_SCOPE_AGENT) <= mygen)
        __builtin_amdgcn_s_sleep(1);
    }
  }
  __syncthreads();
  __builtin_amdgcn_fence(__ATOMIC_ACQUIRE, "agent");   // per-wave L0 invalidate
  mygen++;
}

// -------- weight packing kernels (run once per launch; amortized) --------
__global__ void pack_wencH(const float* __restrict__ W_hh, __bf16* __restrict__ dst) {
  int idx = blockIdx.x * 256 + threadIdx.x;     // 128*16*512 exact
  int frag = idx >> 9, e = idx & 511, lane = e >> 4, p = e & 15;
  int nt = frag >> 4, kt = frag & 15;
  int n = (nt << 4) + (lane & 15);
  int k = (kt << 5) + koff_of(lane, p);
  dst[idx] = (__bf16)W_hh[n * DD + k];
}
__global__ void pack_wencX(const float* __restrict__ W_ih, __bf16* __restrict__ dst) {
  int idx = blockIdx.x * 256 + threadIdx.x;     // 128*11*512 exact
  int frag = idx >> 9, e = idx & 511, lane = e >> 4, p = e & 15;
  int nt = frag / KTX, kt = frag - nt * KTX;
  int n = (nt << 4) + (lane & 15);
  int k = (kt << 5) + koff_of(lane, p);
  dst[idx] = (__bf16)((k < ENCD) ? W_ih[n * ENCD + k] : 0.0f);
}
__global__ void pack_wdec(const float* __restrict__ W_hh, const float* __restrict__ W_ih,
                          const float* __restrict__ W_pred, __bf16* __restrict__ dst) {
  int idx = blockIdx.x * 256 + threadIdx.x;     // 128*16*512 exact
  int frag = idx >> 9, e = idx & 511, lane = e >> 4, p = e & 15;
  int nt = frag >> 4, kt = frag & 15;
  int n = (nt << 4) + (lane & 15);
  int k = (kt << 5) + koff_of(lane, p);
  // W_dec = W_hh + W_ih @ W_pred   (decoder fold, fp32 accumulate)
  float s = W_hh[n * DD + k];
  for (int q = 0; q < ENCD; ++q) s += W_ih[n * ENCD + q] * W_pred[q * DD + k];
  dst[idx] = (__bf16)s;
}
__global__ void pack_wp(const float* __restrict__ W_pred, __bf16* __restrict__ dst) {
  int idx = blockIdx.x * 256 + threadIdx.x;     // 21*16*512 exact
  int frag = idx >> 9, e = idx & 511, lane = e >> 4, p = e & 15;
  int nt = frag >> 4, kt = frag & 15;
  int n = (nt << 4) + (lane & 15);
  int k = (kt << 5) + koff_of(lane, p);
  dst[idx] = (__bf16)((n < ENCD) ? W_pred[n * DD + k] : 0.0f);
}
__global__ void pack_bias(const float* __restrict__ b_ih, const float* __restrict__ b_hh,
                          const float* __restrict__ W_ih, const float* __restrict__ b_pred,
                          float* __restrict__ benc, float* __restrict__ bdec,
                          float* __restrict__ bprd) {
  int g = blockIdx.x * 256 + threadIdx.x;
  if (g < GG) {
    float s = b_ih[g] + b_hh[g];
    benc[g] = s;
    float t = 0.0f;
    for (int q = 0; q < ENCD; ++q) t += W_ih[g * ENCD + q] * b_pred[q];
    bdec[g] = s + t;                 // b_dec = b_ih + b_hh + W_ih @ b_pred
  }
  if (g < PRED) bprd[g] = (g < ENCD) ? b_pred[g] : 0.0f;
}

// -------- per-step helpers --------
__device__ __forceinline__ void stage_x(const float* __restrict__ x_enc, int t,
                                        __bf16* __restrict__ dst, int wg, int tid) {
  // pre-swizzle x_t [128 x 321->352] into A-fragment layout; frags distributed over WGs
  for (int fid = wg; fid < 8 * KTX; fid += NWG) {
    int mt = fid / KTX, kt = fid - mt * KTX;
#pragma unroll
    for (int j = 0; j < 2; ++j) {
      int e = tid + 256 * j;                      // 0..511
      int lane_e = e >> 4, p = e & 15;
      int m = (mt << 4) + (lane_e & 15);
      int k = (kt << 5) + koff_of(lane_e, p);
      float v = (k < ENCD) ? x_enc[((size_t)m * SEQ + t) * ENCD + k] : 0.0f;
      dst[(size_t)fid * 512 + e] = (__bf16)v;
    }
  }
}

__device__ __forceinline__ void gates_update(const v8f* acc, float* creg, const float* bias,
                                             __bf16* __restrict__ hw, int wg, int w, int lane) {
  // this lane holds gate values for d = wg*16 + (lane&15), m = 16w + r + 8*(lane>>4)
  const int kk   = ((wg & 1) << 4) + (lane & 15);    // k-offset of d inside its 32-wide tile
  const int lgrp = 16 * ((kk >> 3) & 1);
  const int p    = (kk & 7) + 8 * (kk >> 4);
  __bf16* base = hw + ((size_t)(w * KTH + (wg >> 1)) * 512) + p;
#pragma unroll
  for (int r = 0; r < 8; ++r) {
    float gi = acc[0][r] + bias[0];
    float gf = acc[1][r] + bias[1];
    float gg = acc[2][r] + bias[2];
    float go = acc[3][r] + bias[3];
    float c  = sigf(gf) * creg[r] + sigf(gi) * tanhf(gg);
    creg[r]  = c;
    float h  = sigf(go) * tanhf(c);
    int L2 = r + 8 * (lane >> 4) + lgrp;            // target lane in A-fragment
    base[L2 * 16] = (__bf16)h;
  }
}

// -------- persistent LSTM kernel --------
__global__ __launch_bounds__(256) void lstm_persistent(
    const float* __restrict__ x_enc, const float* __restrict__ benc,
    const float* __restrict__ bdec, const float* __restrict__ bprd,
    const __bf16* __restrict__ gWencH, const __bf16* __restrict__ gWencX,
    const __bf16* __restrict__ gWdecH, const __bf16* __restrict__ gWp,
    __bf16* __restrict__ hbuf, __bf16* __restrict__ xbuf,
    float* __restrict__ out, unsigned* __restrict__ ctrl) {
  extern __shared__ char smem[];
  __bf16* sWencH = (__bf16*)smem;                   // [4][16][32][16]
  __bf16* sWencX = sWencH + 4 * KTH * 512;          // [4][11][32][16]
  __bf16* sWdecH = sWencX + 4 * KTX * 512;          // [4][16][32][16]
  __bf16* sWp    = sWdecH + 4 * KTH * 512;          // [16][32][16]

  const int wg = blockIdx.x, tid = threadIdx.x;
  const int lane = tid & 31, w = tid >> 5;          // w = M-tile of this wave

  // One-time LDS fill of all step-invariant weights via async-to-LDS DMA path
  for (int q = 0; q < 4; ++q) {
    async_g2l(sWencH + q * KTH * 512, gWencH + (size_t)(q * 32 + wg) * KTH * 512, KTH * 512 * 2);
    async_g2l(sWencX + q * KTX * 512, gWencX + (size_t)(q * 32 + wg) * KTX * 512, KTX * 512 * 2);
    async_g2l(sWdecH + q * KTH * 512, gWdecH + (size_t)(q * 32 + wg) * KTH * 512, KTH * 512 * 2);
  }
  if (wg < NTY) async_g2l(sWp, gWp + (size_t)wg * KTH * 512, KTH * 512 * 2);
  wait_async_all();
  __syncthreads();

  const int n16 = (wg << 4) + (lane & 15);
  float be[4], bd[4];
#pragma unroll
  for (int q = 0; q < 4; ++q) { be[q] = benc[q * DD + n16]; bd[q] = bdec[q * DD + n16]; }

  float creg[8] = {0, 0, 0, 0, 0, 0, 0, 0};         // cell state lives in registers forever
  const v8f vz = {0, 0, 0, 0, 0, 0, 0, 0};

  stage_x(x_enc, 0, xbuf, wg, tid);                 // prestage x_0 (parity 0)
  unsigned mygen = 0;
  grid_sync(ctrl, ctrl + 1, mygen);

  // ---------------- encoder: 720 sequential steps ----------------
  for (int t = 0; t < SEQ; ++t) {
    const int pr = t & 1, pw = pr ^ 1;
    const __bf16* hr = hbuf + pr * H1;
    const __bf16* xr = xbuf + pr * X1;
    v8f acc[4] = {vz, vz, vz, vz};

    for (int kt = 0; kt < KTX; ++kt) {              // x @ W_ih^T slice
      v16bf a = *(const v16bf*)(xr + ((size_t)(w * KTX + kt) * 32 + lane) * 16);
#pragma unroll
      for (int q = 0; q < 4; ++q) {
        v16bf b = *(const v16bf*)(sWencX + ((q * KTX + kt) * 32 + lane) * 16);
        acc[q] = wmma_bf16(a, b, acc[q]);
      }
    }
    for (int kt = 0; kt < KTH; ++kt) {              // h @ W_hh^T slice
      v16bf a = *(const v16bf*)(hr + ((size_t)(w * KTH + kt) * 32 + lane) * 16);
#pragma unroll
      for (int q = 0; q < 4; ++q) {
        v16bf b = *(const v16bf*)(sWencH + ((q * KTH + kt) * 32 + lane) * 16);
        acc[q] = wmma_bf16(a, b, acc[q]);
      }
    }
    if (t + 1 < SEQ) stage_x(x_enc, t + 1, xbuf + pw * X1, wg, tid);  // overlap staging
    gates_update(acc, creg, be, hbuf + pw * H1, wg, w, lane);
    grid_sync(ctrl, ctrl + 1, mygen);
  }

  // ---------------- decoder: 336 autoregressive steps (folded) ----------------
  for (int td = 0; td < PRED; ++td) {
    const int t = SEQ + td;
    const int pr = t & 1, pw = pr ^ 1;
    const __bf16* hr = hbuf + pr * H1;
    v8f acc[4] = {vz, vz, vz, vz};
    v8f accy = vz;

    for (int kt = 0; kt < KTH; ++kt) {
      v16bf a = *(const v16bf*)(hr + ((size_t)(w * KTH + kt) * 32 + lane) * 16);
#pragma unroll
      for (int q = 0; q < 4; ++q) {
        v16bf b = *(const v16bf*)(sWdecH + ((q * KTH + kt) * 32 + lane) * 16);
        acc[q] = wmma_bf16(a, b, acc[q]);
      }
      if (wg < NTY) {                               // yy = h @ W_pred^T (output side GEMM)
        v16bf b = *(const v16bf*)(sWp + (kt * 32 + lane) * 16);
        accy = wmma_bf16(a, b, accy);
      }
    }
    if (wg < NTY) {
      int n = (wg << 4) + (lane & 15);
      if (n < ENCD) {
        float bp = bprd[n];
        int mb = (w << 4) + ((lane >> 4) << 3);
#pragma unroll
        for (int r = 0; r < 8; ++r)
          out[((size_t)(mb + r) * PRED + td) * ENCD + n] = accy[r] + bp;
      }
    }
    gates_update(acc, creg, bd, hbuf + pw * H1, wg, w, lane);
    grid_sync(ctrl, ctrl + 1, mygen);
  }
}

extern "C" void kernel_launch(void* const* d_in, const int* in_sizes, int n_in,
                              void* d_out, int out_size, void* d_ws, size_t ws_size,
                              hipStream_t stream) {
  (void)in_sizes; (void)n_in; (void)out_size; (void)ws_size;
  const float* x_enc  = (const float*)d_in[0];
  const float* W_ih   = (const float*)d_in[1];
  const float* W_hh   = (const float*)d_in[2];
  const float* b_ih   = (const float*)d_in[3];
  const float* b_hh   = (const float*)d_in[4];
  const float* W_pred = (const float*)d_in[5];
  const float* b_pred = (const float*)d_in[6];
  float* out = (float*)d_out;
  char*  ws  = (char*)d_ws;

  unsigned* ctrl  = (unsigned*)(ws + O_CTRL);
  __bf16*   hbuf  = (__bf16*)(ws + O_HBUF);
  __bf16*   xbuf  = (__bf16*)(ws + O_XBUF);
  __bf16*   wencH = (__bf16*)(ws + O_WENCH);
  __bf16*   wencX = (__bf16*)(ws + O_WENCX);
  __bf16*   wdecH = (__bf16*)(ws + O_WDECH);
  __bf16*   wp    = (__bf16*)(ws + O_WP);
  float*    benc  = (float*)(ws + O_BENC);
  float*    bdec  = (float*)(ws + O_BDEC);
  float*    bprd  = (float*)(ws + O_BPRD);

  // zero barrier state + h0 (both parities) every launch (graph-replay safe)
  hipMemsetAsync(ws, 0, O_XBUF, stream);

  pack_wencH<<<4096, 256, 0, stream>>>(W_hh, wencH);
  pack_wencX<<<2816, 256, 0, stream>>>(W_ih, wencX);
  pack_wdec <<<4096, 256, 0, stream>>>(W_hh, W_ih, W_pred, wdecH);
  pack_wp   <<< 672, 256, 0, stream>>>(W_pred, wp);
  pack_bias <<<   8, 256, 0, stream>>>(b_ih, b_hh, W_ih, b_pred, benc, bdec, bprd);

  lstm_persistent<<<NWG, 256, SMEM_BYTES, stream>>>(
      x_enc, benc, bdec, bprd, wencH, wencX, wdecH, wp, hbuf, xbuf, out, ctrl);
}
